// Encoder_15633680957764
// MI455X (gfx1250) — matrix operand order repaired and checked by reading, not compiled, not run on hardware
//
#include <hip/hip_runtime.h>
#include <hip/hip_bf16.h>

typedef _Float16 v16h __attribute__((ext_vector_type(16)));
typedef _Float16 v8h  __attribute__((ext_vector_type(8)));
typedef float    v8f  __attribute__((ext_vector_type(8)));
typedef unsigned int u32x4 __attribute__((ext_vector_type(4)));
typedef int      i32x4 __attribute__((ext_vector_type(4)));
typedef int      i32x8 __attribute__((ext_vector_type(8)));

#define WMMA_F16(a, b, c) \
  __builtin_amdgcn_wmma_f32_16x16x32_f16(false, (a), false, (b), (short)0, (c), false, false)

static __device__ __forceinline__ v16h cat16(v8h lo, v8h hi) {
  return __builtin_shufflevector(lo, hi, 0,1,2,3,4,5,6,7,8,9,10,11,12,13,14,15);
}

// ---------------------------------------------------------------------------
// 16-lane-row reductions via DPP row rotations (VALU only, no LDS traffic).
// ---------------------------------------------------------------------------
template <int CTRL>
static __device__ __forceinline__ float dpp_ror(float v) {
  int r = __builtin_amdgcn_update_dpp(0, __builtin_bit_cast(int, v), CTRL, 0xf, 0xf, true);
  return __builtin_bit_cast(float, r);
}
static __device__ __forceinline__ float rmax16(float v) {
  v = fmaxf(v, dpp_ror<0x128>(v));
  v = fmaxf(v, dpp_ror<0x124>(v));
  v = fmaxf(v, dpp_ror<0x122>(v));
  v = fmaxf(v, dpp_ror<0x121>(v));
  return v;
}
static __device__ __forceinline__ float rsum16(float v) {
  v += dpp_ror<0x128>(v);
  v += dpp_ror<0x124>(v);
  v += dpp_ror<0x122>(v);
  v += dpp_ror<0x121>(v);
  return v;
}

// ---------------------------------------------------------------------------
// TDM: 2D tile (tile_w elems x tile_h rows, row stride in elems) -> LDS.
// Wave-level instruction; caller gates to one wave, tracks with TENSORcnt.
// ---------------------------------------------------------------------------
static __device__ __forceinline__ void tdm_load_2d(unsigned int lds_addr, const void* gaddr,
                                                   unsigned int tile_w, unsigned int tile_h,
                                                   unsigned int stride_elems) {
  unsigned long long ga = (unsigned long long)(uintptr_t)gaddr;
  const unsigned int tdim = 1u << 20;  // OOB dims: large (all accesses in-bounds)
  u32x4 g0;
  g0[0] = 1u;                                                     // count=1 valid user D#
  g0[1] = lds_addr;                                               // LDS byte address
  g0[2] = (unsigned int)ga;                                       // global_addr[31:0]
  g0[3] = (unsigned int)((ga >> 32) & 0x01FFFFFFu) | 0x80000000u; // addr[56:32] | type=2
  i32x8 g1;
  g1[0] = 0x00010000;                                             // data_size=1 (2B), no flags
  g1[1] = (int)((tdim & 0xffffu) << 16);                          // tensor_dim0[15:0]
  g1[2] = (int)(((tdim >> 16) & 0xffffu) | ((tdim & 0xffffu) << 16));
  g1[3] = (int)(((tdim >> 16) & 0xffffu) | (tile_w << 16));       // tile_dim0
  g1[4] = (int)(tile_h & 0xffffu);                                // tile_dim1 (tile_dim2=0)
  g1[5] = (int)stride_elems;                                      // tensor_dim0_stride[31:0]
  g1[6] = 0;
  g1[7] = 0;
  i32x4 gz = {0, 0, 0, 0};
#if defined(__clang_major__) && __clang_major__ >= 23
  i32x8 gz8 = {0, 0, 0, 0, 0, 0, 0, 0};
  __builtin_amdgcn_tensor_load_to_lds(g0, g1, gz, gz, gz8, 0);
#else
  __builtin_amdgcn_tensor_load_to_lds(g0, g1, gz, gz, 0);
#endif
}

// ---------------------------------------------------------------------------
// Transposing f32 -> f16 weight conversion: in[K][N] -> outT[N][K].
// 32x32 LDS tile, coalesced loads and stores. Grid: (N/32, K/32), 256 thr.
// ---------------------------------------------------------------------------
__global__ __launch_bounds__(256)
void cvtT_kernel(const float* __restrict__ in, _Float16* __restrict__ outT, int K, int N) {
  __shared__ float tile[32][33];
  const int n0 = blockIdx.x * 32, k0 = blockIdx.y * 32;
  const int tx = threadIdx.x & 31, ty = threadIdx.x >> 5;  // ty: 0..7
#pragma unroll
  for (int i = 0; i < 4; ++i)
    tile[ty + 8 * i][tx] = in[(size_t)(k0 + ty + 8 * i) * N + n0 + tx];
  __syncthreads();
#pragma unroll
  for (int i = 0; i < 4; ++i)
    outT[(size_t)(n0 + ty + 8 * i) * K + k0 + tx] = (_Float16)tile[tx][ty + 8 * i];
}

// ---------------------------------------------------------------------------
// Block reduction helper (256 threads)
// ---------------------------------------------------------------------------
static __device__ __forceinline__ float block_reduce_sum(float v, float* red) {
  int t = threadIdx.x;
  red[t] = v;
  __syncthreads();
  for (int s = 128; s > 0; s >>= 1) {
    if (t < s) red[t] += red[t + s];
    __syncthreads();
  }
  float r = red[0];
  __syncthreads();
  return r;
}

// ---------------------------------------------------------------------------
// LN1: h = LN(x) -> f16   (one block per row, D=768)
// ---------------------------------------------------------------------------
__global__ __launch_bounds__(256)
void ln1_kernel(const float* __restrict__ x, const float* __restrict__ g,
                const float* __restrict__ b, _Float16* __restrict__ h) {
  __shared__ float red[256];
  const int row = blockIdx.x, t = threadIdx.x;
  const float* xr = x + (size_t)row * 768;
  float v[3];
#pragma unroll
  for (int i = 0; i < 3; ++i) v[i] = xr[t + i * 256];
  float mu = block_reduce_sum(v[0] + v[1] + v[2], red) * (1.0f / 768.0f);
  float sq = 0.f;
#pragma unroll
  for (int i = 0; i < 3; ++i) { float d = v[i] - mu; sq += d * d; }
  float var = block_reduce_sum(sq, red) * (1.0f / 768.0f);
  float inv = rsqrtf(var + 1e-5f);
#pragma unroll
  for (int i = 0; i < 3; ++i) {
    int c = t + i * 256;
    h[(size_t)row * 768 + c] = (_Float16)((v[i] - mu) * inv * g[c] + b[c]);
  }
}

// ---------------------------------------------------------------------------
// x1 = x + msa(reshaped); h2 = LN2(x1) -> f16
// ---------------------------------------------------------------------------
__global__ __launch_bounds__(256)
void addln2_kernel(const float* __restrict__ x, const _Float16* __restrict__ msa,
                   const float* __restrict__ g, const float* __restrict__ b,
                   float* __restrict__ x1, _Float16* __restrict__ h2) {
  __shared__ float red[256];
  const int row = blockIdx.x, t = threadIdx.x;
  const int bb = row >> 10, nn = row & 1023;
  float v[3];
#pragma unroll
  for (int i = 0; i < 3; ++i) {
    int c = t + i * 256;
    int hh = c >> 6, dd = c & 63;
    float m = (float)msa[((((size_t)bb * 12 + hh) << 10) + nn) * 64 + dd];
    v[i] = x[(size_t)row * 768 + c] + m;
    x1[(size_t)row * 768 + c] = v[i];
  }
  float mu = block_reduce_sum(v[0] + v[1] + v[2], red) * (1.0f / 768.0f);
  float sq = 0.f;
#pragma unroll
  for (int i = 0; i < 3; ++i) { float d = v[i] - mu; sq += d * d; }
  float var = block_reduce_sum(sq, red) * (1.0f / 768.0f);
  float inv = rsqrtf(var + 1e-5f);
#pragma unroll
  for (int i = 0; i < 3; ++i) {
    int c = t + i * 256;
    h2[(size_t)row * 768 + c] = (_Float16)((v[i] - mu) * inv * g[c] + b[c]);
  }
}

// ---------------------------------------------------------------------------
// f16 WMMA GEMM: C(MxN) = A(MxK,row) @ B, with B given TRANSPOSED: Bt[N][K].
// Block tile 64x256, 8 waves (2x4), wave tile 32x64 (2x4 frags), BK=32.
// Both A and Bt tiles stream via TDM (tensor_load_to_lds), double-buffered.
// EPI 0: scatter f16 to [B,H,N,64], value*scale        (Q/K projections)
// EPI 1: f16 gelu(v + bias[n]) row-major               (FC1)
// EPI 2: f32 v + bias[n] + resid[m,n] row-major        (FC2 + residual)
// EPI 3: scatter f16 to [B,H,64,N] (transposed V)      (V projection)
// ---------------------------------------------------------------------------
template <int EPI>
__global__ __launch_bounds__(256)
void gemm_f16_kernel(const _Float16* __restrict__ A, const _Float16* __restrict__ Bt,
                     int M, int N, int K,
                     _Float16* __restrict__ out16, float* __restrict__ outf,
                     const float* __restrict__ bias, const float* __restrict__ resid,
                     float scale) {
  __shared__ _Float16 As[2][64 * 32];   // [m][k], TDM-filled
  __shared__ _Float16 Bs[2][256 * 32];  // [n][k], TDM-filled (Bt is row-major [N][K])
  const int t = threadIdx.x;
  const int lane = t & 31, w = t >> 5;
  const int half = lane >> 4, ln16 = lane & 15;
  const int wm = w >> 2, wn = w & 3;
  const int m0 = blockIdx.y * 64, n0 = blockIdx.x * 256;
  const int nK = K / 32;

  v8f c[2][4] = {};

  if (w == 0) {
    tdm_load_2d((unsigned int)(uintptr_t)&As[0][0], A + (size_t)m0 * K, 32, 64, (unsigned int)K);
    tdm_load_2d((unsigned int)(uintptr_t)&Bs[0][0], Bt + (size_t)n0 * K, 32, 256, (unsigned int)K);
    __builtin_amdgcn_s_wait_tensorcnt(0);
  }
  __syncthreads();

  for (int kt = 0; kt < nK; ++kt) {
    const int cur = kt & 1, nxt = cur ^ 1;
    if ((kt + 1) < nK && w == 0) {
      tdm_load_2d((unsigned int)(uintptr_t)&As[nxt][0],
                  A + (size_t)m0 * K + (kt + 1) * 32, 32, 64, (unsigned int)K);
      tdm_load_2d((unsigned int)(uintptr_t)&Bs[nxt][0],
                  Bt + (size_t)n0 * K + (kt + 1) * 32, 32, 256, (unsigned int)K);
    }

    v16h af[2], bf[4];
#pragma unroll
    for (int i = 0; i < 2; ++i) {
      int r = wm * 32 + i * 16 + ln16;
      af[i] = *reinterpret_cast<const v16h*>(&As[cur][r * 32 + 16 * half]);
    }
#pragma unroll
    for (int j = 0; j < 4; ++j) {
      int n = wn * 64 + j * 16 + ln16;
      bf[j] = *reinterpret_cast<const v16h*>(&Bs[cur][n * 32 + 16 * half]);
    }
#pragma unroll
    for (int i = 0; i < 2; ++i)
#pragma unroll
      for (int j = 0; j < 4; ++j)
        c[i][j] = WMMA_F16(af[i], bf[j], c[i][j]);

    if (w == 0) __builtin_amdgcn_s_wait_tensorcnt(0);
    __syncthreads();
  }

#pragma unroll
  for (int i = 0; i < 2; ++i)
#pragma unroll
    for (int j = 0; j < 4; ++j)
#pragma unroll
      for (int r = 0; r < 8; ++r) {
        int m = m0 + wm * 32 + i * 16 + r + 8 * half;
        int n = n0 + wn * 64 + j * 16 + ln16;
        if (m >= M || n >= N) continue;
        float v = c[i][j][r];
        if (EPI == 0) {
          int bb = m >> 10, nn = m & 1023, hh = n >> 6, dd = n & 63;
          out16[((((size_t)bb * 12 + hh) << 10) + nn) * 64 + dd] = (_Float16)(v * scale);
        } else if (EPI == 1) {
          float xg = v + bias[n];
          float gl = 0.5f * xg * (1.0f + erff(xg * 0.70710678118654752f));
          out16[(size_t)m * N + n] = (_Float16)gl;
        } else if (EPI == 2) {
          outf[(size_t)m * N + n] = v + bias[n] + resid[(size_t)m * N + n];
        } else {
          int bb = m >> 10, nn = m & 1023, hh = n >> 6, dd = n & 63;
          out16[(((size_t)bb * 12 + hh) * 64 + dd) * 1024 + nn] = (_Float16)v;  // V^T
        }
      }
}

// ---------------------------------------------------------------------------
// Flash attention: one block = (bh, 64-query block); 4 waves x 16 query rows.
// Q pre-scaled by 1/8. V supplied transposed [B,H,64,N]. K/V tiles stream via
// TDM, double-buffered. msa output f16 in [B,H,N,64].
// ---------------------------------------------------------------------------
__global__ __launch_bounds__(128)
void attn_kernel(const _Float16* __restrict__ q, const _Float16* __restrict__ k,
                 const _Float16* __restrict__ vt, _Float16* __restrict__ msa) {
  __shared__ _Float16 Qs[64 * 64];      // [qr][d]
  __shared__ _Float16 Ks[2][32 * 64];   // [j][d]  == B^T for S = Q K^T (TDM)
  __shared__ _Float16 Vt[2][64 * 32];   // [d][j]  == B^T for O = P V   (TDM)
  __shared__ _Float16 Ps[4 * 16 * 32];  // per-wave P tile

  const int t = threadIdx.x, lane = t & 31, w = t >> 5;
  const int half = lane >> 4, ln16 = lane & 15;
  const int bh = blockIdx.y, qb = blockIdx.x;
  const _Float16* qp = q + (size_t)bh * 1024 * 64;
  const _Float16* kp = k + (size_t)bh * 1024 * 64;
  const _Float16* vp = vt + (size_t)bh * 1024 * 64;  // [64][1024]
  _Float16* mp = msa + (size_t)bh * 1024 * 64;

  {  // load Q tile 64x64 (one-time)
    int qr = t >> 1, c = (t & 1) * 32;
    const uint4* gp = reinterpret_cast<const uint4*>(qp + (size_t)(qb * 64 + qr) * 64 + c);
    uint4* sp = reinterpret_cast<uint4*>(&Qs[qr * 64 + c]);
#pragma unroll
    for (int i = 0; i < 4; ++i) sp[i] = gp[i];
  }
  if (w == 0) {
    tdm_load_2d((unsigned int)(uintptr_t)&Ks[0][0], kp, 64, 32, 64);
    tdm_load_2d((unsigned int)(uintptr_t)&Vt[0][0], vp, 32, 64, 1024);
    __builtin_amdgcn_s_wait_tensorcnt(0);
  }
  __syncthreads();

  v16h aQ[2];
#pragma unroll
  for (int ks = 0; ks < 2; ++ks) {
    int m = w * 16 + ln16;
    aQ[ks] = *reinterpret_cast<const v16h*>(&Qs[m * 64 + ks * 32 + 16 * half]);
  }

  v8f o[4] = {};
  float mrow[8], lrow[8];
#pragma unroll
  for (int r = 0; r < 8; ++r) { mrow[r] = -1e30f; lrow[r] = 0.f; }

  for (int jb = 0; jb < 32; ++jb) {
    const int cur = jb & 1, nxt = cur ^ 1;
    if ((jb + 1) < 32 && w == 0) {
      tdm_load_2d((unsigned int)(uintptr_t)&Ks[nxt][0], kp + (size_t)(jb + 1) * 32 * 64, 64, 32, 64);
      tdm_load_2d((unsigned int)(uintptr_t)&Vt[nxt][0], vp + (size_t)(jb + 1) * 32, 32, 64, 1024);
    }

    v8f s0 = {}, s1 = {};
#pragma unroll
    for (int ks = 0; ks < 2; ++ks) {
      v16h b0 = *reinterpret_cast<const v16h*>(&Ks[cur][(ln16) * 64 + ks * 32 + 16 * half]);
      v16h b1 = *reinterpret_cast<const v16h*>(&Ks[cur][(16 + ln16) * 64 + ks * 32 + 16 * half]);
      s0 = WMMA_F16(aQ[ks], b0, s0);
      s1 = WMMA_F16(aQ[ks], b1, s1);
    }

#pragma unroll
    for (int r = 0; r < 8; ++r) {
      float tmax = rmax16(fmaxf(s0[r], s1[r]));
      float nm = fmaxf(mrow[r], tmax);
      float sc = __expf(mrow[r] - nm);
      mrow[r] = nm;
      float p0 = __expf(s0[r] - nm);
      float p1 = __expf(s1[r] - nm);
      lrow[r] = lrow[r] * sc + rsum16(p0 + p1);
#pragma unroll
      for (int jf = 0; jf < 4; ++jf) o[jf][r] *= sc;
      int base = w * 512 + (r + 8 * half) * 32;
      Ps[base + ln16] = (_Float16)p0;
      Ps[base + 16 + ln16] = (_Float16)p1;
    }

    {  // O += P(16x32) @ V(32x64)
      v16h aP = *reinterpret_cast<const v16h*>(&Ps[w * 512 + ln16 * 32 + 16 * half]);
#pragma unroll
      for (int jf = 0; jf < 4; ++jf) {
        v16h bv = *reinterpret_cast<const v16h*>(&Vt[cur][(jf * 16 + ln16) * 32 + 16 * half]);
        o[jf] = WMMA_F16(aP, bv, o[jf]);
      }
    }
    if (w == 0) __builtin_amdgcn_s_wait_tensorcnt(0);
    __syncthreads();
  }

#pragma unroll
  for (int r = 0; r < 8; ++r) {
    float inv = 1.0f / lrow[r];
    int nq = qb * 64 + w * 16 + r + 8 * half;
#pragma unroll
    for (int jf = 0; jf < 4; ++jf) {
      int d = jf * 16 + ln16;
      mp[(size_t)nq * 64 + d] = (_Float16)(o[jf][r] * inv);
    }
  }
}

// ---------------------------------------------------------------------------
extern "C" void kernel_launch(void* const* d_in, const int* in_sizes, int n_in,
                              void* d_out, int out_size, void* d_ws, size_t ws_size,
                              hipStream_t stream) {
  (void)in_sizes; (void)n_in; (void)out_size; (void)ws_size;
  const float* x    = (const float*)d_in[0];
  const float* wq   = (const float*)d_in[1];
  const float* wk   = (const float*)d_in[2];
  const float* wv   = (const float*)d_in[3];
  const float* ln1g = (const float*)d_in[4];
  const float* ln1b = (const float*)d_in[5];
  const float* ln2g = (const float*)d_in[6];
  const float* ln2b = (const float*)d_in[7];
  const float* fc1w = (const float*)d_in[8];
  const float* fc1b = (const float*)d_in[9];
  const float* fc2w = (const float*)d_in[10];
  const float* fc2b = (const float*)d_in[11];
  float* out = (float*)d_out;

  const int M = 8192, D = 768, F = 3072;
  char* wsp = (char*)d_ws;
  auto alloc = [&](size_t bytes) {
    char* p = wsp;
    wsp += (bytes + 255) & ~(size_t)255;
    return p;
  };
  _Float16* wq16T  = (_Float16*)alloc((size_t)D * D * 2);  // [D][D] transposed
  _Float16* wk16T  = (_Float16*)alloc((size_t)D * D * 2);
  _Float16* wv16T  = (_Float16*)alloc((size_t)D * D * 2);
  _Float16* fc1T   = (_Float16*)alloc((size_t)D * F * 2);  // [F][D]
  _Float16* fc2T   = (_Float16*)alloc((size_t)F * D * 2);  // [D][F]
  _Float16* h1     = (_Float16*)alloc((size_t)M * D * 2);
  _Float16* q16    = (_Float16*)alloc((size_t)M * D * 2);  // [B,H,N,64]
  _Float16* k16    = (_Float16*)alloc((size_t)M * D * 2);  // [B,H,N,64]
  _Float16* v16t   = (_Float16*)alloc((size_t)M * D * 2);  // [B,H,64,N]
  _Float16* msa16  = (_Float16*)alloc((size_t)M * D * 2);  // [B,H,N,64]
  float*    x1f    = (float*)alloc((size_t)M * D * 4);
  _Float16* h2     = (_Float16*)alloc((size_t)M * D * 2);
  _Float16* mid16  = (_Float16*)alloc((size_t)M * F * 2);

  // transposing weight conversions: in[K][N] -> outT[N][K]
  cvtT_kernel<<<dim3(D / 32, D / 32), 256, 0, stream>>>(wq, wq16T, D, D);
  cvtT_kernel<<<dim3(D / 32, D / 32), 256, 0, stream>>>(wk, wk16T, D, D);
  cvtT_kernel<<<dim3(D / 32, D / 32), 256, 0, stream>>>(wv, wv16T, D, D);
  cvtT_kernel<<<dim3(F / 32, D / 32), 256, 0, stream>>>(fc1w, fc1T, D, F);
  cvtT_kernel<<<dim3(D / 32, F / 32), 256, 0, stream>>>(fc2w, fc2T, F, D);

  ln1_kernel<<<M, 256, 0, stream>>>(x, ln1g, ln1b, h1);

  dim3 gQKV(D / 256, M / 64);
  gemm_f16_kernel<0><<<gQKV, 256, 0, stream>>>(h1, wq16T, M, D, D, q16, nullptr, nullptr, nullptr, 0.125f);
  gemm_f16_kernel<0><<<gQKV, 256, 0, stream>>>(h1, wk16T, M, D, D, k16, nullptr, nullptr, nullptr, 1.0f);
  gemm_f16_kernel<3><<<gQKV, 256, 0, stream>>>(h1, wv16T, M, D, D, v16t, nullptr, nullptr, nullptr, 1.0f);

  attn_kernel<<<dim3(16, 96), 128, 0, stream>>>(q16, k16, v16t, msa16);

  addln2_kernel<<<M, 256, 0, stream>>>(x, msa16, ln2g, ln2b, x1f, h2);

  gemm_f16_kernel<1><<<dim3(F / 256, M / 64), 256, 0, stream>>>(h2, fc1T, M, F, D, mid16, nullptr, fc1b, nullptr, 1.0f);
  gemm_f16_kernel<2><<<dim3(D / 256, M / 64), 256, 0, stream>>>(mid16, fc2T, M, D, F, nullptr, out, fc2b, x1f, 1.0f);
}
